// Qwen3MoeExperts_85933705658459
// MI455X (gfx1250) — compile-verified, matrix operand order checked by reading, not run on hardware
//
#include <hip/hip_runtime.h>
#include <hip/hip_bf16.h>
#include <stdint.h>

// ---------------------------------------------------------------------------
// Qwen3 MoE single-expert MLP on gfx1250 (MI455X).
//   h   = silu(H @ G^T) * (H @ U^T)   [32768 x 768]
//   out = h @ D^T                     [32768 x 1024]
// bf16 WMMA (v_wmma_f32_16x16x32_bf16) with f32 accumulation.
// Kernel 1: fp32->bf16 on the staging path (v_cvt_pk_bf16_f32 / v_perm_b32).
// Kernel 2: A tile is raw bf16 -> copied global->LDS with
//           global_load_async_to_lds_b128 (ASYNCcnt / s_wait_asynccnt path).
// Staging uses per-thread base pointers advanced by BK per iteration so the
// per-pass deltas fold into load/store immediate offsets.
// ---------------------------------------------------------------------------

constexpr int kHidden = 1024;
constexpr int kInter  = 768;
constexpr int kTokens = 32768;

constexpr int BM  = 128;  // token rows per block
constexpr int BK  = 64;   // k per LDS stage == 2 bf16-WMMA K-steps
constexpr int LDT = 72;   // LDS row stride in bf16 elems (64 + 8 pad, 144B)

typedef __bf16 bf16x16 __attribute__((ext_vector_type(16)));
typedef float  f32x8   __attribute__((ext_vector_type(8)));

union Frag {
  bf16x16 v;
  uint4   q[2];
};

// ---- fp32 -> packed bf16x2 ----
#if defined(__has_builtin) && __has_builtin(__builtin_amdgcn_cvt_pk_bf16_f32)
typedef __bf16 bf16x2 __attribute__((ext_vector_type(2)));
__device__ __forceinline__ uint32_t pk_bf16(float lo, float hi) {
  union { bf16x2 v; uint32_t u; } cv;
  cv.v = __builtin_amdgcn_cvt_pk_bf16_f32(lo, hi);
  return cv.u;
}
#else
// 2x v_add + 1x v_perm_b32: rounded-to-nearest-even bf16 pair.
__device__ __forceinline__ uint32_t pk_bf16(float lo, float hi) {
  uint32_t a = __float_as_uint(lo);
  uint32_t b = __float_as_uint(hi);
  a += 0x7FFFu + ((a >> 16) & 1u);
  b += 0x7FFFu + ((b >> 16) & 1u);
  // dst = { a[23:16]->b0, a[31:24]->b1, b[23:16]->b2, b[31:24]->b3 }
  return __builtin_amdgcn_perm(b, a, 0x07060302u);
}
#endif

__device__ __forceinline__ uint2 pk4(const float4& v) {
  uint2 r;
  r.x = pk_bf16(v.x, v.y);
  r.y = pk_bf16(v.z, v.w);
  return r;
}

__device__ __forceinline__ uint16_t bf16_1(float f) {
  uint32_t a = __float_as_uint(f);
  a += 0x7FFFu + ((a >> 16) & 1u);
  return (uint16_t)(a >> 16);
}

// ---- gfx1250 async global->LDS copy (16B per lane), ASYNCcnt tracked ----
__device__ __forceinline__ void async_copy_b128(uint32_t lds_off, const void* g) {
  unsigned long long ga = (unsigned long long)(uintptr_t)g;
  asm volatile("global_load_async_to_lds_b128 %0, %1, off"
               :: "v"(lds_off), "v"(ga)
               : "memory");
}

__device__ __forceinline__ void wait_async0() {
#if defined(__has_builtin) && __has_builtin(__builtin_amdgcn_s_wait_asynccnt)
  __builtin_amdgcn_s_wait_asynccnt(0);
#else
  asm volatile("s_wait_asynccnt 0x0" ::: "memory");
#endif
}

// Per-lane WMMA fragment load from a padded LDS tile (K-step ks in {0,1}).
__device__ __forceinline__ Frag ld_frag(const uint16_t* s, int row, int kh, int ks) {
  Frag f;
  const uint16_t* p = s + row * LDT + ks * 32 + kh * 8;
  f.q[0] = *(const uint4*)(p);
  f.q[1] = *(const uint4*)(p + 16);
  return f;
}

// ---------------------------------------------------------------------------
// Kernel 1: fused gate+up GEMM + SiLU*up epilogue -> bf16 h in workspace.
// Block tile 128(M) x 64(N) x 64(K-stage); 8 waves as 4x2, wave tile 32x32
// for BOTH gate and up (16 WMMAs per wave per stage).
// ---------------------------------------------------------------------------
__global__ __launch_bounds__(256)
void moe_gate_up_kernel(const float* __restrict__ H,
                        const float* __restrict__ Gw,
                        const float* __restrict__ Uw,
                        const int* __restrict__ eidx,
                        uint16_t* __restrict__ hws) {
  constexpr int BN = 64;
  __shared__ __align__(16) uint16_t sA[BM * LDT];
  __shared__ __align__(16) uint16_t sG[BN * LDT];
  __shared__ __align__(16) uint16_t sU[BN * LDT];

  const int t    = threadIdx.x;
  const int lane = t & 31;
  const int wave = t >> 5;
  const int wm   = wave >> 1;   // 0..3 -> M offset wm*32
  const int wn   = wave & 1;    // 0..1 -> N offset wn*32
  const int bn   = blockIdx.x * BN;
  const int bm   = blockIdx.y * BM;

  const size_t eoff = (size_t)eidx[0] * (size_t)kInter * (size_t)kHidden;

  const int rA = t >> 4;   // 0..15 rows per staging pass
  const int cA = t & 15;   // float4 chunk within a 64-wide K row

  // Per-thread base pointers; per-pass deltas become immediate offsets.
  const float* pH = H + (size_t)(bm + rA) * kHidden + cA * 4;
  const float* pG = Gw + eoff + (size_t)(bn + rA) * kHidden + cA * 4;
  const float* pU = Uw + eoff + (size_t)(bn + rA) * kHidden + cA * 4;
  uint16_t* tA = &sA[rA * LDT + cA * 4];
  uint16_t* tG = &sG[rA * LDT + cA * 4];
  uint16_t* tU = &sU[rA * LDT + cA * 4];

  f32x8 accG[2][2] = {};
  f32x8 accU[2][2] = {};

  const int mrow = lane & 15;
  const int kh   = lane >> 4;

  for (int bk = 0; bk < kHidden; bk += BK) {
    // ---- stage tiles into LDS with fp32 -> bf16 conversion ----
#pragma unroll
    for (int p = 0; p < 8; ++p) {
      const float4 v = *(const float4*)(pH + p * 16 * kHidden);
      *(uint2*)(tA + p * 16 * LDT) = pk4(v);
    }
#pragma unroll
    for (int p = 0; p < 4; ++p) {
      const float4 g = *(const float4*)(pG + p * 16 * kHidden);
      *(uint2*)(tG + p * 16 * LDT) = pk4(g);
      const float4 u = *(const float4*)(pU + p * 16 * kHidden);
      *(uint2*)(tU + p * 16 * LDT) = pk4(u);
    }
    if (bk + BK < kHidden) {  // gfx1250 global_prefetch_b8 on next K tile
      __builtin_prefetch(pH + BK, 0, 0);
      __builtin_prefetch(pG + BK, 0, 0);
      __builtin_prefetch(pU + BK, 0, 0);
    }
    pH += BK; pG += BK; pU += BK;
    __syncthreads();

#pragma unroll
    for (int ks = 0; ks < 2; ++ks) {
      Frag af[2], gf[2], uf[2];
#pragma unroll
      for (int i = 0; i < 2; ++i) {
        af[i] = ld_frag(sA, wm * 32 + i * 16 + mrow, kh, ks);
        gf[i] = ld_frag(sG, wn * 32 + i * 16 + mrow, kh, ks);
        uf[i] = ld_frag(sU, wn * 32 + i * 16 + mrow, kh, ks);
      }
#pragma unroll
      for (int mt = 0; mt < 2; ++mt)
#pragma unroll
        for (int nt = 0; nt < 2; ++nt) {
          accG[mt][nt] = __builtin_amdgcn_wmma_f32_16x16x32_bf16(
              false, af[mt].v, false, gf[nt].v, (short)0, accG[mt][nt], false, false);
          accU[mt][nt] = __builtin_amdgcn_wmma_f32_16x16x32_bf16(
              false, af[mt].v, false, uf[nt].v, (short)0, accU[mt][nt], false, false);
        }
    }
    __syncthreads();
  }

  // ---- epilogue: h = silu(g) * u, store bf16 ----
  const int mbase = bm + wm * 32 + (lane >> 4) * 8;
  const int nbase = bn + wn * 32 + (lane & 15);
#pragma unroll
  for (int mt = 0; mt < 2; ++mt)
#pragma unroll
    for (int nt = 0; nt < 2; ++nt) {
#pragma unroll
      for (int rr = 0; rr < 8; ++rr) {
        const float g = accG[mt][nt][rr];
        const float u = accU[mt][nt][rr];
        const float s = g / (1.0f + __expf(-g));   // SiLU
        const int m = mbase + mt * 16 + rr;
        const int n = nbase + nt * 16;
        hws[(size_t)m * kInter + n] = bf16_1(s * u);
      }
    }
}

// ---------------------------------------------------------------------------
// Kernel 2: out = h @ D^T. Block tile 128(M) x 128(N) x 64(K-stage);
// 8 waves as 4x2, wave tile 32x64 (16 WMMAs per wave per stage).
// A tile (bf16 h) copied via global_load_async_to_lds_b128.
// ---------------------------------------------------------------------------
__global__ __launch_bounds__(256)
void moe_down_kernel(const uint16_t* __restrict__ hws,
                     const float* __restrict__ Dw,
                     const int* __restrict__ eidx,
                     float* __restrict__ out) {
  constexpr int BN = 128;
  __shared__ __align__(16) uint16_t sA[BM * LDT];
  __shared__ __align__(16) uint16_t sD[BN * LDT];

  const int t    = threadIdx.x;
  const int lane = t & 31;
  const int wave = t >> 5;
  const int wm   = wave >> 1;   // 0..3 -> M offset wm*32
  const int wn   = wave & 1;    // 0..1 -> N offset wn*64
  const int bn   = blockIdx.x * BN;   // HIDDEN cols
  const int bm   = blockIdx.y * BM;   // token rows

  const int rA = t >> 4;   // 0..15 (D staging)
  const int cA = t & 15;   // float4 chunk
  const int r8 = t >> 3;   // 0..31 (async A copy)
  const int c8 = t & 7;    // 16B chunk, 8 per 64-elem row

  // Base addresses, advanced by BK per iteration.
  const uint16_t* pA = hws + (size_t)(bm + r8) * kInter + c8 * 8;
  const float*    pD = Dw + (size_t)eidx[0] * (size_t)kHidden * (size_t)kInter
                          + (size_t)(bn + rA) * kInter + cA * 4;
  const uint32_t  lA = (uint32_t)(uintptr_t)&sA[r8 * LDT + c8 * 8];
  uint16_t*       tD = &sD[rA * LDT + cA * 4];

  f32x8 acc[2][4] = {};

  const int mrow = lane & 15;
  const int kh   = lane >> 4;

  for (int bk = 0; bk < kInter; bk += BK) {
    // ---- A tile: raw bf16, async global->LDS (no VGPR round-trip) ----
#pragma unroll
    for (int p = 0; p < 4; ++p)
      async_copy_b128(lA + p * 32 * LDT * 2, pA + p * 32 * kInter);
    // ---- D tile: fp32 -> bf16 staging ----
#pragma unroll
    for (int p = 0; p < 8; ++p) {
      const float4 d = *(const float4*)(pD + p * 16 * kInter);
      *(uint2*)(tD + p * 16 * LDT) = pk4(d);
    }
    if (bk + BK < kInter) {
      __builtin_prefetch(pA + BK, 0, 0);
      __builtin_prefetch(pD + BK, 0, 0);
    }
    pA += BK; pD += BK;
    wait_async0();
    __syncthreads();

#pragma unroll
    for (int ks = 0; ks < 2; ++ks) {
      Frag af[2], df[4];
#pragma unroll
      for (int i = 0; i < 2; ++i)
        af[i] = ld_frag(sA, wm * 32 + i * 16 + mrow, kh, ks);
#pragma unroll
      for (int j = 0; j < 4; ++j)
        df[j] = ld_frag(sD, wn * 64 + j * 16 + mrow, kh, ks);
#pragma unroll
      for (int mt = 0; mt < 2; ++mt)
#pragma unroll
        for (int nt = 0; nt < 4; ++nt)
          acc[mt][nt] = __builtin_amdgcn_wmma_f32_16x16x32_bf16(
              false, af[mt].v, false, df[nt].v, (short)0, acc[mt][nt], false, false);
    }
    __syncthreads();
  }

  const int mbase = bm + wm * 32 + (lane >> 4) * 8;
  const int nbase = bn + wn * 64 + (lane & 15);
#pragma unroll
  for (int mt = 0; mt < 2; ++mt)
#pragma unroll
    for (int nt = 0; nt < 4; ++nt) {
#pragma unroll
      for (int rr = 0; rr < 8; ++rr) {
        const int m = mbase + mt * 16 + rr;
        const int n = nbase + nt * 16;
        out[(size_t)m * kHidden + n] = acc[mt][nt][rr];
      }
    }
}

// ---------------------------------------------------------------------------
// Launch: inputs = {hidden_states, gate_proj, up_proj, down_proj, expert_idx}
// ---------------------------------------------------------------------------
extern "C" void kernel_launch(void* const* d_in, const int* in_sizes, int n_in,
                              void* d_out, int out_size, void* d_ws, size_t ws_size,
                              hipStream_t stream) {
  (void)in_sizes; (void)n_in; (void)out_size; (void)ws_size;

  const float* H  = (const float*)d_in[0];
  const float* Gw = (const float*)d_in[1];
  const float* Uw = (const float*)d_in[2];
  const float* Dw = (const float*)d_in[3];
  const int* eidx = (const int*)d_in[4];

  uint16_t* hws = (uint16_t*)d_ws;     // bf16 intermediate h: 32768*768*2 B
  float* out    = (float*)d_out;

  dim3 blk(256, 1, 1);
  dim3 g1(kInter / 64, kTokens / BM, 1);    // 12 x 256
  moe_gate_up_kernel<<<g1, blk, 0, stream>>>(H, Gw, Uw, eidx, hws);

  dim3 g2(kHidden / 128, kTokens / BM, 1);  // 8 x 256
  moe_down_kernel<<<g2, blk, 0, stream>>>(hws, Dw, eidx, out);
}
